// AVWDCRNN_62852551410347
// MI455X (gfx1250) — matrix operand order repaired and checked by reading, not compiled, not run on hardware
//
#include <hip/hip_runtime.h>
#include <hip/hip_bf16.h>

// ---------------------------------------------------------------------------
// CDNA5 WMMA types / helpers
// ---------------------------------------------------------------------------
typedef __attribute__((ext_vector_type(16))) _Float16 v16h;
typedef __attribute__((ext_vector_type(8)))  float    v8f;

__device__ __forceinline__ v8f wmma32(v16h a, v16h b, v8f c) {
  return __builtin_amdgcn_wmma_f32_16x16x32_f16(
      /*neg_a=*/false, a, /*neg_b=*/false, b,
      /*c_mod=*/(short)0, c, /*reuse_a=*/false, /*reuse_b=*/false);
}

// Fragments are stored pre-swizzled: one 16x32(A) or 32x16(B) tile = 512 f16
// = 1024B block; lane reads its 16 halves contiguously (2x global_load_b128).
__device__ __forceinline__ v16h load_frag(const _Float16* __restrict__ p) {
  return *(const v16h*)(p + (size_t)(threadIdx.x & 31) * 16);
}

// ---------------------------------------------------------------------------
// Constants
// ---------------------------------------------------------------------------
#define NN   307
#define NP   320     // padded N (10 K-chunks of 32, 20 M-tiles of 16)
#define BB   64
#define TT   12
#define DE   16
#define KSUP 5

// ---------------------------------------------------------------------------
// Supports: sup[0]=I, sup[1]=A=softmax(relu(E E^T)), sup[3]=adj_n (row-norm)
// ---------------------------------------------------------------------------
__global__ void build_supports_a(const float* __restrict__ E,
                                 const float* __restrict__ adj,
                                 float* __restrict__ sup) {
  int n = blockIdx.x;
  int tid = threadIdx.x;  // 128
  __shared__ float en[DE];
  __shared__ float srow[NN];
  __shared__ float red[128];
  if (tid < DE) en[tid] = E[n * DE + tid];
  __syncthreads();
  for (int m = tid; m < NN; m += 128) {
    float s = 0.f;
#pragma unroll
    for (int d = 0; d < DE; ++d) s += en[d] * E[m * DE + d];
    srow[m] = s > 0.f ? s : 0.f;
  }
  __syncthreads();
  float mx = -1e30f;
  for (int m = tid; m < NN; m += 128) mx = fmaxf(mx, srow[m]);
  red[tid] = mx; __syncthreads();
  for (int s = 64; s > 0; s >>= 1) { if (tid < s) red[tid] = fmaxf(red[tid], red[tid + s]); __syncthreads(); }
  mx = red[0]; __syncthreads();
  float sum = 0.f;
  for (int m = tid; m < NN; m += 128) sum += __expf(srow[m] - mx);
  red[tid] = sum; __syncthreads();
  for (int s = 64; s > 0; s >>= 1) { if (tid < s) red[tid] += red[tid + s]; __syncthreads(); }
  float inv = 1.f / red[0]; __syncthreads();
  float asum = 0.f;
  for (int m = tid; m < NN; m += 128) asum += adj[n * NN + m];
  red[tid] = asum; __syncthreads();
  for (int s = 64; s > 0; s >>= 1) { if (tid < s) red[tid] += red[tid + s]; __syncthreads(); }
  float ainv = 1.f / (red[0] + 1e-8f);
  for (int m = tid; m < NN; m += 128) {
    sup[0 * NN * NN + n * NN + m] = (m == n) ? 1.f : 0.f;
    sup[1 * NN * NN + n * NN + m] = __expf(srow[m] - mx) * inv;
    sup[3 * NN * NN + n * NN + m] = adj[n * NN + m] * ainv;
  }
}

// sup[2] = 2*A@A - I ; sup[4] = adj_n @ adj_n   (tiny 307^3 matmuls, VALU)
__global__ void build_supports_b(float* __restrict__ sup) {
  int col = blockIdx.x * 16 + threadIdx.x;
  int row = blockIdx.y * 16 + threadIdx.y;
  int mode = blockIdx.z;
  if (row >= NN || col >= NN) return;
  const float* S = sup + (size_t)(mode == 0 ? 1 : 3) * NN * NN;
  float acc = 0.f;
  for (int k = 0; k < NN; ++k) acc += S[row * NN + k] * S[k * NN + col];
  if (mode == 0) sup[2 * NN * NN + row * NN + col] = 2.f * acc - ((row == col) ? 1.f : 0.f);
  else           sup[4 * NN * NN + row * NN + col] = acc;
}

// supports f32 -> pre-swizzled A-fragment f16 blocks  [5][mt=20][kt=10][512]
__global__ void sup_to_frag(const float* __restrict__ sup, _Float16* __restrict__ suphA) {
  int tid  = threadIdx.x;         // 0..511 = lane*16 + j
  int tile = blockIdx.x;          // 0..199 = mt*10 + kt
  int k    = blockIdx.y;          // 0..4
  int mt = tile / 10, kt = tile % 10;
  int lane = tid >> 4, j = tid & 15;
  int m   = mt * 16 + (lane & 15);
  int kkd = kt * 32 + ((lane >> 4) << 4) + j;
  float v = (m < NN && kkd < NN) ? sup[((size_t)k * NN + m) * NN + kkd] : 0.f;
  suphA[(((size_t)k * 20 + mt) * 10 + kt) * 512 + tid] = (_Float16)v;
}

// ---------------------------------------------------------------------------
// Per-node weights into B-fragment layout: [n][nt=O/16][kt=J/32][512]
// ---------------------------------------------------------------------------
__global__ void node_weights(const float* __restrict__ E, const float* __restrict__ W,
                             _Float16* __restrict__ WnB, int J, int O) {
  int n = blockIdx.y;
  long idx = (long)blockIdx.x * blockDim.x + threadIdx.x;
  long JO = (long)J * O;
  if (idx >= JO) return;
  int jj = (int)(idx / O), o = (int)(idx % O);
  float acc = 0.f;
#pragma unroll
  for (int d = 0; d < DE; ++d) acc += E[n * DE + d] * W[(long)d * JO + idx];
  int KT = J >> 5, NT = O >> 4;
  int kt = jj >> 5, kr = jj & 31, nt = o >> 4, nc = o & 15;
  int lane = ((kr >> 4) << 4) + nc, j = kr & 15;
  WnB[(((size_t)n * NT + nt) * KT + kt) * 512 + lane * 16 + j] = (_Float16)acc;
}

__global__ void node_bias(const float* __restrict__ E, const float* __restrict__ b,
                          float* __restrict__ bn, int O) {
  int n = blockIdx.x, o = threadIdx.x;
  float acc = 0.f;
#pragma unroll
  for (int d = 0; d < DE; ++d) acc += E[n * DE + d] * b[d * O + o];
  bn[n * O + o] = acc;
}

// ---------------------------------------------------------------------------
// Window-average + concat into B-fragment layouts.
//   X1B [b][nt=12][kt=10][512] : cols 0:64 = xa_t, 64:128 = h, 128:192 = c
//   X2B [b][nt=8 ][kt=10][512] : cols 0:64 = xa_t (64:128 filled by gate)
// Rows n>=307 zero-padded.
// ---------------------------------------------------------------------------
__global__ void concat_kernel(const float* __restrict__ xin, const float* __restrict__ h,
                              const float* __restrict__ c, _Float16* __restrict__ X1B,
                              _Float16* __restrict__ X2B, int t) {
  int ch = threadIdx.x;   // 0..191
  int n  = blockIdx.x;    // 0..319
  int b  = blockIdx.y;    // 0..63
  float v = 0.f;
  if (n < NN) {
    if (ch < 64) {
      size_t st = (size_t)NN * 64;
      size_t base = ((size_t)b * TT) * st + (size_t)n * 64 + ch;
      if (t == 0 || t == TT - 1) v = xin[base + (size_t)t * st];
      else v = (xin[base + (size_t)(t - 1) * st] + xin[base + (size_t)t * st] +
                xin[base + (size_t)(t + 1) * st]) * (1.f / 3.f);
    } else if (ch < 128) {
      v = h[((size_t)b * NN + n) * 64 + (ch - 64)];
    } else {
      v = c[((size_t)b * NN + n) * 64 + (ch - 128)];
    }
  }
  int kt = n >> 5, kr = n & 31;
  int nt = ch >> 4, nc = ch & 15;
  int lane = ((kr >> 4) << 4) + nc, j = kr & 15;
  X1B[(((size_t)b * 12 + nt) * 10 + kt) * 512 + lane * 16 + j] = (_Float16)v;
  if (ch < 64)
    X2B[(((size_t)b * 8 + nt) * 10 + kt) * 512 + lane * 16 + j] = (_Float16)v;
  else if (n >= NN && ch < 128)
    X2B[(((size_t)b * 8 + nt) * 10 + kt) * 512 + lane * 16 + j] = (_Float16)0.f;
}

// ---------------------------------------------------------------------------
// gconv stage 1 (WMMA): x_g[b,k,n,c] = sum_m sup[k][n][m] * X[b][m][c]
// One wave per 16x16 tile, K=320 (10 chunks). Output written directly in
// A-fragment layout for the per-node GEMM:  XcatA[n][mtb=4][ktj=KC/32][512].
// ---------------------------------------------------------------------------
__global__ void gconv1(const _Float16* __restrict__ suphA, const _Float16* __restrict__ XB,
                       _Float16* __restrict__ XcatA, int C, int KC) {
  int wave = threadIdx.x >> 5;                 // 0..7
  int b = blockIdx.y, k = blockIdx.z;
  int NCT = C >> 4;
  int tile = blockIdx.x * 8 + wave;            // grid.x*8 == 20*NCT exactly
  int mt = tile / NCT, ct = tile % NCT;
  const _Float16* Abase = suphA + (((size_t)k * 20 + mt) * 10) * 512;
  const _Float16* Bbase = XB + (((size_t)b * NCT + ct) * 10) * 512;
  v8f acc = {};
#pragma unroll
  for (int kk = 0; kk < 10; ++kk) {
    if (kk < 9) __builtin_prefetch(Bbase + (kk + 1) * 512, 0, 0);
    v16h a  = load_frag(Abase + kk * 512);
    v16h bf = load_frag(Bbase + kk * 512);
    acc = wmma32(a, bf, acc);
  }
  int lane = threadIdx.x & 31;
  int cn = ct * 16 + (lane & 15);
  int rbase = mt * 16 + ((lane >> 4) << 3);
  int jj = k * C + cn;
  int KTj = KC >> 5;
  int ktj = jj >> 5, kk2 = jj & 31;
  int lanet = ((kk2 >> 4) << 4) + (b & 15);
  int jp = kk2 & 15;
  size_t base = ((size_t)(b >> 4) * KTj + ktj) * 512 + lanet * 16 + jp;
  size_t nstride = (size_t)4 * KTj * 512;
#pragma unroll
  for (int r = 0; r < 8; ++r)
    XcatA[(size_t)(rbase + r) * nstride + base] = (_Float16)acc[r];
}

// ---------------------------------------------------------------------------
// Gate GEMM per node n: (64x960)x(960x128)+bias; epilogue: z->zbuf, r*h->X2B.
// 8 waves; wave = o-tile; each wave accumulates 4 b-tiles.
// ---------------------------------------------------------------------------
__global__ void gate_kernel(const _Float16* __restrict__ XcatA, const _Float16* __restrict__ WnB,
                            const float* __restrict__ bn, const float* __restrict__ h,
                            float* __restrict__ zbuf, _Float16* __restrict__ X2B) {
  const int KT = 30, O = 128;
  int n = blockIdx.x;
  int w = threadIdx.x >> 5;
  int lane = threadIdx.x & 31;
  v8f acc0 = {}, acc1 = {}, acc2 = {}, acc3 = {};
  const _Float16* Ab = XcatA + (size_t)n * 4 * KT * 512;
  const _Float16* Bb = WnB + ((size_t)n * 8 + w) * KT * 512;
  for (int kk = 0; kk < KT; ++kk) {
    v16h bf = load_frag(Bb + kk * 512);
    v16h a0 = load_frag(Ab + (0 * KT + kk) * 512); acc0 = wmma32(a0, bf, acc0);
    v16h a1 = load_frag(Ab + (1 * KT + kk) * 512); acc1 = wmma32(a1, bf, acc1);
    v16h a2 = load_frag(Ab + (2 * KT + kk) * 512); acc2 = wmma32(a2, bf, acc2);
    v16h a3 = load_frag(Ab + (3 * KT + kk) * 512); acc3 = wmma32(a3, bf, acc3);
  }
  int o = w * 16 + (lane & 15);
  float bias = bn[n * O + o];
  int rb = (lane >> 4) << 3;
  int kt2 = n >> 5, kr2 = n & 31;
  int nt2 = o >> 4, nc2 = o & 15;
  int lane2 = ((kr2 >> 4) << 4) + nc2, j2 = kr2 & 15;
  v8f accs[4] = {acc0, acc1, acc2, acc3};
#pragma unroll
  for (int bt = 0; bt < 4; ++bt) {
#pragma unroll
    for (int r = 0; r < 8; ++r) {
      int b = bt * 16 + rb + r;
      float g = 1.f / (1.f + __expf(-(accs[bt][r] + bias)));
      if (o < 64) {
        zbuf[((size_t)b * NN + n) * 64 + o] = g;
      } else {
        float hv = h[((size_t)b * NN + n) * 64 + (o - 64)];
        X2B[(((size_t)b * 8 + nt2) * 10 + kt2) * 512 + lane2 * 16 + j2] = (_Float16)(g * hv);
      }
    }
  }
}

// ---------------------------------------------------------------------------
// Candidate GEMM per node (64x640)x(640x64)+bias, tanh, GRU/HiPPO update.
// 4 waves; wave = o-tile.
// ---------------------------------------------------------------------------
__global__ void cand_kernel(const _Float16* __restrict__ Xcat2A, const _Float16* __restrict__ WnB,
                            const float* __restrict__ bn, const float* __restrict__ zbuf,
                            float* __restrict__ h, float* __restrict__ c,
                            float* __restrict__ out, int t) {
  const int KT = 20, O = 64;
  int n = blockIdx.x;
  int w = threadIdx.x >> 5;
  int lane = threadIdx.x & 31;
  v8f acc0 = {}, acc1 = {}, acc2 = {}, acc3 = {};
  const _Float16* Ab = Xcat2A + (size_t)n * 4 * KT * 512;
  const _Float16* Bb = WnB + ((size_t)n * 4 + w) * KT * 512;
  for (int kk = 0; kk < KT; ++kk) {
    v16h bf = load_frag(Bb + kk * 512);
    v16h a0 = load_frag(Ab + (0 * KT + kk) * 512); acc0 = wmma32(a0, bf, acc0);
    v16h a1 = load_frag(Ab + (1 * KT + kk) * 512); acc1 = wmma32(a1, bf, acc1);
    v16h a2 = load_frag(Ab + (2 * KT + kk) * 512); acc2 = wmma32(a2, bf, acc2);
    v16h a3 = load_frag(Ab + (3 * KT + kk) * 512); acc3 = wmma32(a3, bf, acc3);
  }
  int o = w * 16 + (lane & 15);
  float bias = bn[n * O + o];
  int rb = (lane >> 4) << 3;
  float tf = (float)t;
  float alpha = tf / (tf + 1.f), beta = 1.f / (tf + 1.f);
  v8f accs[4] = {acc0, acc1, acc2, acc3};
#pragma unroll
  for (int bt = 0; bt < 4; ++bt) {
#pragma unroll
    for (int r = 0; r < 8; ++r) {
      int b = bt * 16 + rb + r;
      float hc = tanhf(accs[bt][r] + bias);
      size_t idx = ((size_t)b * NN + n) * 64 + o;
      float z = zbuf[idx];
      float hn = z * h[idx] + (1.f - z) * hc;
      h[idx] = hn;
      c[idx] = alpha * c[idx] + beta * hn;
      out[(((size_t)b * TT + t) * NN + n) * 64 + o] = hn;
    }
  }
}

// ---------------------------------------------------------------------------
// Host launch
// ---------------------------------------------------------------------------
extern "C" void kernel_launch(void* const* d_in, const int* in_sizes, int n_in,
                              void* d_out, int out_size, void* d_ws, size_t ws_size,
                              hipStream_t stream) {
  (void)in_sizes; (void)n_in; (void)out_size; (void)ws_size;
  const float* x          = (const float*)d_in[0];
  const float* init_state = (const float*)d_in[1];
  const float* init_c     = (const float*)d_in[2];
  const float* E          = (const float*)d_in[3];
  const float* adj        = (const float*)d_in[4];
  const float* Wg         = (const float*)d_in[5];
  const float* bg         = (const float*)d_in[6];
  const float* Wc         = (const float*)d_in[7];
  const float* bc         = (const float*)d_in[8];
  float* out = (float*)d_out;

  char* ws = (char*)d_ws;
  size_t off = 0;
  auto alloc = [&](size_t bytes) -> char* {
    char* p = ws + off; off += (bytes + 1023) & ~(size_t)1023; return p;
  };
  float*    sup    = (float*)   alloc((size_t)KSUP * NN * NN * 4);
  _Float16* suphA  = (_Float16*)alloc((size_t)KSUP * 20 * 10 * 512 * 2);
  _Float16* WngB   = (_Float16*)alloc((size_t)NN * 8 * 30 * 512 * 2);
  _Float16* WncB   = (_Float16*)alloc((size_t)NN * 4 * 20 * 512 * 2);
  float*    bgn    = (float*)   alloc((size_t)NN * 128 * 4);
  float*    bcn    = (float*)   alloc((size_t)NN * 64 * 4);
  _Float16* X1B    = (_Float16*)alloc((size_t)BB * 12 * 10 * 512 * 2);
  _Float16* X2B    = (_Float16*)alloc((size_t)BB * 8 * 10 * 512 * 2);
  _Float16* XcatA  = (_Float16*)alloc((size_t)NP * 4 * 30 * 512 * 2);
  _Float16* Xcat2A = (_Float16*)alloc((size_t)NP * 4 * 20 * 512 * 2);
  float*    zb     = (float*)   alloc((size_t)BB * NN * 64 * 4);
  float*    hb     = (float*)   alloc((size_t)BB * NN * 64 * 4);
  float*    cb     = (float*)   alloc((size_t)BB * NN * 64 * 4);
  float*    cur0   = (float*)   alloc((size_t)BB * TT * NN * 64 * 4);

  build_supports_a<<<NN, 128, 0, stream>>>(E, adj, sup);
  build_supports_b<<<dim3(20, 20, 2), dim3(16, 16), 0, stream>>>(sup);
  sup_to_frag<<<dim3(200, KSUP), 512, 0, stream>>>(sup, suphA);

  const size_t HS = (size_t)BB * NN * 64 * sizeof(float);
  for (int l = 0; l < 2; ++l) {
    const float* lin = (l == 0) ? x : cur0;
    float* lout = (l == 0) ? cur0 : out;
    node_weights<<<dim3((960 * 128) / 256, NN), 256, 0, stream>>>(
        E, Wg + (size_t)l * DE * 960 * 128, WngB, 960, 128);
    node_weights<<<dim3((640 * 64) / 256, NN), 256, 0, stream>>>(
        E, Wc + (size_t)l * DE * 640 * 64, WncB, 640, 64);
    node_bias<<<NN, 128, 0, stream>>>(E, bg + (size_t)l * DE * 128, bgn, 128);
    node_bias<<<NN, 64, 0, stream>>>(E, bc + (size_t)l * DE * 64, bcn, 64);
    hipMemcpyAsync(hb, init_state + (size_t)l * BB * NN * 64, HS, hipMemcpyDeviceToDevice, stream);
    hipMemcpyAsync(cb, init_c     + (size_t)l * BB * NN * 64, HS, hipMemcpyDeviceToDevice, stream);
    for (int t = 0; t < TT; ++t) {
      concat_kernel<<<dim3(NP, BB), 192, 0, stream>>>(lin, hb, cb, X1B, X2B, t);
      gconv1<<<dim3(30, BB, KSUP), 256, 0, stream>>>(suphA, X1B, XcatA, 192, 960);
      gate_kernel<<<NN, 256, 0, stream>>>(XcatA, WngB, bgn, hb, zb, X2B);
      gconv1<<<dim3(20, BB, KSUP), 256, 0, stream>>>(suphA, X2B, Xcat2A, 128, 640);
      cand_kernel<<<NN, 128, 0, stream>>>(Xcat2A, WncB, bcn, zb, hb, cb, lout, t);
    }
  }
}
